// GATv2Model_77824807403686
// MI455X (gfx1250) — matrix operand order repaired and checked by reading, not compiled, not run on hardware
//
#include <hip/hip_runtime.h>
#include <hip/hip_bf16.h>

#define DD 256
#define NN 64
#define BB 64
#define HH 8
#define CC 32
#define LL 4
#define NEG_SLOPE 0.2f

typedef __attribute__((ext_vector_type(16))) __bf16 v16bf;
typedef __attribute__((ext_vector_type(8)))  float  v8f;
typedef __attribute__((ext_vector_type(4)))  unsigned int u32x4;

union FragBF {
    v16bf v;
    u32x4 q[2];
};

// ---------------- reduction helpers (wave32) ----------------
__device__ __forceinline__ float wave_reduce_sum(float v) {
    #pragma unroll
    for (int o = 16; o > 0; o >>= 1) v += __shfl_xor(v, o, 32);
    return v;
}
__device__ __forceinline__ float wave_reduce_max(float v) {
    #pragma unroll
    for (int o = 16; o > 0; o >>= 1) v = fmaxf(v, __shfl_xor(v, o, 32));
    return v;
}
// 256-thread block reduce; red must have >= 8 floats of LDS
__device__ __forceinline__ float block_reduce_sum(float v, float* red) {
    const int lane = threadIdx.x & 31;
    v = wave_reduce_sum(v);
    __syncthreads();
    if (lane == 0) red[threadIdx.x >> 5] = v;
    __syncthreads();
    float r = (lane < 8) ? red[lane] : 0.f;
    r = wave_reduce_sum(r);   // every wave reduces the 8 partials -> all threads hold total
    return r;
}

// ---------------- f32 -> bf16 convert ----------------
__global__ void __launch_bounds__(256)
f32_to_bf16_kernel(const float* __restrict__ src, __hip_bfloat16* __restrict__ dst, int n) {
    for (int i = blockIdx.x * blockDim.x + threadIdx.x; i < n; i += gridDim.x * blockDim.x)
        dst[i] = __float2bfloat16(src[i]);
}

// ---------------- edge embedding: gather + max-norm-1 clip -> bf16 ----------------
__global__ void __launch_bounds__(256)
edge_embed_kernel(const int* __restrict__ cats, const float* __restrict__ table,
                  __hip_bfloat16* __restrict__ ebf) {
    __shared__ float red[8];
    const int row = blockIdx.x;           // one edge per block, 256 threads = D
    const int d   = threadIdx.x;
    const float v = table[cats[row] * DD + d];
    const float ss = block_reduce_sum(v * v, red);
    const float nrm = sqrtf(ss);
    const float scale = (nrm > 1.f) ? (1.f / nrm) : 1.f;   // min(1, 1/nrm); nrm==0 -> 1
    ebf[(size_t)row * DD + d] = __float2bfloat16(v * scale);
}

// ---------------- input Linear(2 -> D) (K too small for WMMA) ----------------
__global__ void __launch_bounds__(256)
mlp1_kernel(const float* __restrict__ x, const float* __restrict__ W,
            const float* __restrict__ b, float* __restrict__ z, int total) {
    for (int i = blockIdx.x * blockDim.x + threadIdx.x; i < total; i += gridDim.x * blockDim.x) {
        const int m = i >> 8, d = i & 255;
        z[i] = fmaf(W[d * 2], x[m * 2], fmaf(W[d * 2 + 1], x[m * 2 + 1], b[d]));
    }
}

// ---------------- LayerNorm (+optional relu, +optional residual) -> f32 & bf16 ----------------
__global__ void __launch_bounds__(256)
ln_kernel(const float* __restrict__ z, const float* __restrict__ g, const float* __restrict__ be,
          const float* __restrict__ res, float* __restrict__ out_f32,
          __hip_bfloat16* __restrict__ out_bf, int relu) {
    __shared__ float red[8];
    const int row = blockIdx.x;
    const int d   = threadIdx.x;
    const size_t o = (size_t)row * DD + d;
    const float v = z[o];
    const float mean = block_reduce_sum(v, red) * (1.f / DD);
    const float dv = v - mean;
    const float var = block_reduce_sum(dv * dv, red) * (1.f / DD);
    float y = dv * rsqrtf(var + 1e-5f) * g[d] + be[d];
    if (relu) y = fmaxf(y, 0.f);
    if (res)  y += res[o];
    out_f32[o] = y;
    if (out_bf) out_bf[o] = __float2bfloat16(y);
}

// ---------------- WMMA GEMM: C[M,N] = A[M,K=256](bf16) @ W[N,K]^T(bf16) + bias ----------------
// block = 256 threads = 8 waves; block tile = 128x64; wave tile = 16x64 (4 WMMA chains).
// The 64x256 weight tile (contiguous 32 KB) is staged once per block in LDS; B fragments
// then come from ds_load_b128, A fragments from global (prefetched while B is staged).
__global__ void __launch_bounds__(256)
gemm_bf16_wmma(const __hip_bfloat16* __restrict__ A, const __hip_bfloat16* __restrict__ W,
               const float* __restrict__ bias, float* __restrict__ C,
               int M, int Nn, int K /* must be 256 */) {
    __shared__ __hip_bfloat16 Bs[64 * 256];   // 32 KB weight tile

    const int lane = threadIdx.x & 31;
    const int wave = threadIdx.x >> 5;
    const int mBase = blockIdx.y * 128 + wave * 16;
    const int nBase = blockIdx.x * 64;
    const int r   = lane & 15;
    const int kha = (lane >> 4) * 8;    // A-fragment K sub-offset for this half-wave
    const int khb = (lane >> 4) * 8;    // B-fragment K sub-offset for this half-wave

    const __hip_bfloat16* Arow = A + (size_t)(mBase + r) * 256;

    // warm the A rows for this wave while the block stages B (global_prefetch_b8)
    __builtin_prefetch((const void*)(Arow), 0, 3);
    __builtin_prefetch((const void*)(Arow + 128), 0, 3);

    // cooperative stage of the contiguous 64x256 bf16 weight tile
    {
        const u32x4* src = (const u32x4*)(W + (size_t)nBase * 256);
        u32x4* dst = (u32x4*)Bs;
        #pragma unroll
        for (int i = 0; i < (64 * 256 / 8) / 256; ++i)
            dst[threadIdx.x + i * 256] = src[threadIdx.x + i * 256];
    }
    __syncthreads();

    v8f acc[4] = {v8f{}, v8f{}, v8f{}, v8f{}};
    #pragma unroll
    for (int k0 = 0; k0 < 256; k0 += 32) {
        FragBF a;
        // A 16x32: lane<16 -> K in [0,8)+[16,24); lane>=16 -> [8,16)+[24,32)
        a.q[0] = *(const u32x4*)(Arow + k0 + kha);
        a.q[1] = *(const u32x4*)(Arow + k0 + kha + 16);
        #pragma unroll
        for (int ct = 0; ct < 4; ++ct) {
            // B 32x16: lane<16 -> K in [0,8)+[16,24); lane>=16 -> [8,16)+[24,32) (from LDS)
            const __hip_bfloat16* Brow = Bs + (ct * 16 + r) * 256;
            FragBF b;
            b.q[0] = *(const u32x4*)(Brow + k0 + khb);
            b.q[1] = *(const u32x4*)(Brow + k0 + khb + 16);
            acc[ct] = __builtin_amdgcn_wmma_f32_16x16x32_bf16(false, a.v, false, b.v,
                                                              (short)0, acc[ct], false, false);
        }
    }

    const int rowB = mBase + 8 * (lane >> 4);  // C/D layout: VGPR v -> M = v + 8*(lane>=16)
    #pragma unroll
    for (int ct = 0; ct < 4; ++ct) {
        const int col = nBase + ct * 16 + (lane & 15);
        const float bi = bias ? bias[col] : 0.f;
        #pragma unroll
        for (int v = 0; v < 8; ++v)
            C[(size_t)(rowB + v) * Nn + col] = acc[ct][v] + bi;
    }
}

// ---------------- fused GATv2 attention: scores + softmax(src) + aggregate ----------------
// grid = (N tgt, B), block = 256 threads = 8 waves; wave w == head w, lane == channel c
__global__ void __launch_bounds__(256)
gat_attn_kernel(const float* __restrict__ xl, const float* __restrict__ xr,
                const float* __restrict__ ee, const float* __restrict__ att,
                const float* __restrict__ cb, float* __restrict__ m_f32,
                __hip_bfloat16* __restrict__ m_bf) {
    __shared__ __hip_bfloat16 xls[NN * DD];   // 32 KB: xl[b] staged once, read twice
    __shared__ float sc[HH * NN];             // 2 KB scores

    const int tgt = blockIdx.x;
    const int b   = blockIdx.y;
    const int d   = threadIdx.x;              // d = h*32 + c
    const int h   = d >> 5;
    const int c   = d & 31;

    const float* xlb = xl + (size_t)b * NN * DD;
    for (int i = d; i < NN * DD; i += 256) xls[i] = __float2bfloat16(xlb[i]);
    __syncthreads();

    const float xr_d  = xr[((size_t)b * NN + tgt) * DD + d];
    const float att_d = att[d];

    // scores: s[src,h] = sum_c att * leakyrelu(xl + xr + ee)
    for (int src = 0; src < NN; ++src) {
        float t = __bfloat162float(xls[src * DD + d]) + xr_d
                + ee[((size_t)src * NN + tgt) * DD + d];
        t = (t > 0.f) ? t : NEG_SLOPE * t;
        float p = wave_reduce_sum(att_d * t);
        if (c == 0) sc[h * NN + src] = p;
    }
    __syncthreads();

    // softmax over src per head: wave h owns its 64 scores, 2 per lane
    float v0 = sc[h * NN + c];
    float v1 = sc[h * NN + c + 32];
    const float mx  = wave_reduce_max(fmaxf(v0, v1));
    const float e0  = __expf(v0 - mx);
    const float e1  = __expf(v1 - mx);
    const float inv = 1.f / wave_reduce_sum(e0 + e1);
    sc[h * NN + c]      = e0 * inv;
    sc[h * NN + c + 32] = e1 * inv;
    __syncthreads();

    // aggregate: m[d] = sum_src a[src,h] * xl[b,src,d]
    float acc = 0.f;
    for (int src = 0; src < NN; ++src)
        acc = fmaf(sc[h * NN + src], __bfloat162float(xls[src * DD + d]), acc);
    acc += cb[d];
    const size_t o = ((size_t)b * NN + tgt) * DD + d;
    m_f32[o] = acc;
    m_bf[o]  = __float2bfloat16(acc);
}

// ---------------- output projection: out[m,3] = h[m,:] @ Wo^T + bo ----------------
__global__ void __launch_bounds__(256)
outproj_kernel(const float* __restrict__ h, const float* __restrict__ Wo,
               const float* __restrict__ bo, float* __restrict__ out) {
    __shared__ float red[8];
    const int m = blockIdx.x;
    const int d = threadIdx.x;
    const float hv = h[(size_t)m * DD + d];
    #pragma unroll
    for (int j = 0; j < 3; ++j) {
        const float p = block_reduce_sum(hv * Wo[j * DD + d], red);
        if (d == 0) out[m * 3 + j] = p + bo[j];
    }
}

// =====================================================================
extern "C" void kernel_launch(void* const* d_in, const int* in_sizes, int n_in,
                              void* d_out, int out_size, void* d_ws, size_t ws_size,
                              hipStream_t stream) {
    const float* x    = (const float*)d_in[0];
    const int*   ecat = (const int*)  d_in[1];
    const float* etab = (const float*)d_in[2];
    const float* Wi1  = (const float*)d_in[3];
    const float* bi1  = (const float*)d_in[4];
    const float* g1   = (const float*)d_in[5];
    const float* be1  = (const float*)d_in[6];
    const float* Wi2  = (const float*)d_in[7];
    const float* bi2  = (const float*)d_in[8];
    const float* g2   = (const float*)d_in[9];
    const float* be2  = (const float*)d_in[10];
    const float* Wl   = (const float*)d_in[11];
    const float* bl   = (const float*)d_in[12];
    const float* Wr   = (const float*)d_in[13];
    const float* br   = (const float*)d_in[14];
    const float* We   = (const float*)d_in[15];
    const float* att  = (const float*)d_in[16];
    const float* cb   = (const float*)d_in[17];
    const float* Wp   = (const float*)d_in[18];
    const float* bp   = (const float*)d_in[19];
    const float* lg   = (const float*)d_in[20];
    const float* lb   = (const float*)d_in[21];
    const float* Wo   = (const float*)d_in[22];
    const float* bo   = (const float*)d_in[23];
    float* out = (float*)d_out;

    const int M = BB * NN;       // 4096 node-rows
    const int E = NN * NN;       // 4096 edges
    const int WSZ = DD * DD;     // 65536 weights per matrix

    // bump allocator over workspace
    char* p = (char*)d_ws;
    auto alloc = [&](size_t bytes) -> void* {
        void* r = (void*)p;
        p += (bytes + 255) & ~(size_t)255;
        return r;
    };
    __hip_bfloat16* e_bf  = (__hip_bfloat16*)alloc((size_t)E * DD * 2);
    float*          z     = (float*)         alloc((size_t)M * DD * 4);
    float*          h     = (float*)         alloc((size_t)M * DD * 4);
    __hip_bfloat16* h_bf  = (__hip_bfloat16*)alloc((size_t)M * DD * 2);
    float*          xlb   = (float*)         alloc((size_t)M * DD * 4);
    float*          xrb   = (float*)         alloc((size_t)M * DD * 4);
    float*          eeb   = (float*)         alloc((size_t)E * DD * 4);
    float*          mbuf  = (float*)         alloc((size_t)M * DD * 4);
    __hip_bfloat16* m_bf  = (__hip_bfloat16*)alloc((size_t)M * DD * 2);
    __hip_bfloat16* Wi2b  = (__hip_bfloat16*)alloc((size_t)WSZ * 2);
    __hip_bfloat16* Wlb   = (__hip_bfloat16*)alloc((size_t)LL * WSZ * 2);
    __hip_bfloat16* Wrb   = (__hip_bfloat16*)alloc((size_t)LL * WSZ * 2);
    __hip_bfloat16* Web   = (__hip_bfloat16*)alloc((size_t)LL * WSZ * 2);
    __hip_bfloat16* Wpb   = (__hip_bfloat16*)alloc((size_t)LL * WSZ * 2);
    (void)ws_size; (void)in_sizes; (void)n_in; (void)out_size;

    // ---- weight conversion to bf16 (deterministic, every call) ----
    f32_to_bf16_kernel<<<256, 256, 0, stream>>>(Wi2, Wi2b, WSZ);
    f32_to_bf16_kernel<<<512, 256, 0, stream>>>(Wl,  Wlb,  LL * WSZ);
    f32_to_bf16_kernel<<<512, 256, 0, stream>>>(Wr,  Wrb,  LL * WSZ);
    f32_to_bf16_kernel<<<512, 256, 0, stream>>>(We,  Web,  LL * WSZ);
    f32_to_bf16_kernel<<<512, 256, 0, stream>>>(Wp,  Wpb,  LL * WSZ);

    // ---- edge embedding ----
    edge_embed_kernel<<<E, 256, 0, stream>>>(ecat, etab, e_bf);

    // ---- input MLP ----
    mlp1_kernel<<<1024, 256, 0, stream>>>(x, Wi1, bi1, z, M * DD);
    ln_kernel<<<M, 256, 0, stream>>>(z, g1, be1, nullptr, h, h_bf, 1);          // LN + relu
    gemm_bf16_wmma<<<dim3(DD / 64, M / 128), 256, 0, stream>>>(h_bf, Wi2b, bi2, z, M, DD, DD);
    ln_kernel<<<M, 256, 0, stream>>>(z, g2, be2, nullptr, h, h_bf, 0);          // LN

    // ---- GATv2 layers ----
    for (int l = 0; l < LL; ++l) {
        const size_t wo = (size_t)l * WSZ;
        gemm_bf16_wmma<<<dim3(DD / 64, M / 128), 256, 0, stream>>>(h_bf, Wlb + wo, bl + l * DD, xlb, M, DD, DD);
        gemm_bf16_wmma<<<dim3(DD / 64, M / 128), 256, 0, stream>>>(h_bf, Wrb + wo, br + l * DD, xrb, M, DD, DD);
        gemm_bf16_wmma<<<dim3(DD / 64, E / 128), 256, 0, stream>>>(e_bf, Web + wo, nullptr, eeb, E, DD, DD);
        gat_attn_kernel<<<dim3(NN, BB), 256, 0, stream>>>(xlb, xrb, eeb,
                                                          att + l * HH * CC, cb + l * DD,
                                                          mbuf, m_bf);
        gemm_bf16_wmma<<<dim3(DD / 64, M / 128), 256, 0, stream>>>(m_bf, Wpb + wo, bp + l * DD, z, M, DD, DD);
        ln_kernel<<<M, 256, 0, stream>>>(z, lg + l * DD, lb + l * DD, h, h, h_bf, 1); // relu(LN)+residual
    }

    // ---- output projection ----
    outproj_kernel<<<M, 256, 0, stream>>>(h, Wo, bo, out);
}
